// LLaDAModel_21285857919735
// MI455X (gfx1250) — compile-verified, matrix operand order checked by reading, not run on hardware
//
#include <hip/hip_runtime.h>
#include <hip/hip_bf16.h>

typedef unsigned short u16;
typedef unsigned int   u32;

typedef __attribute__((ext_vector_type(16))) __bf16 v16bf;
typedef __attribute__((ext_vector_type(8)))  float  v8f;
typedef __attribute__((ext_vector_type(4)))  unsigned int v4u;
typedef __attribute__((ext_vector_type(8)))  int    v8i;
typedef __attribute__((ext_vector_type(4)))  int    v4i;

// ---------------- constants (shapes from reference) ----------------
#define CB   8
#define CS   2048
#define CD   1024
#define CH   2048
#define CA   128
#define CE   8
#define CBS  (CB*CS)        // 16384 tokens

#define TILE   128
#define TK     32
#define LDSW   40           // 32 halves + 8 pad (80B rows -> conflict-free b128)
#define TILE_HALVES (TILE * LDSW)      // 5120 halves = 10240 B per tile

#if defined(__has_builtin)
#if __has_builtin(__builtin_amdgcn_tensor_load_to_lds)
#define HAVE_TDM 1
#endif
#endif
#ifndef HAVE_TDM
#define HAVE_TDM 0
#endif

// ---------------- helpers ----------------
__device__ __forceinline__ u16 f2bf(float x) {
    u32 u = __float_as_uint(x);
    u32 r = u + 0x7FFFu + ((u >> 16) & 1u);
    return (u16)(r >> 16);
}
__device__ __forceinline__ float bf2f(u16 h) {
    return __uint_as_float(((u32)h) << 16);
}
__device__ __forceinline__ float silu(float x) {
    return x / (1.f + __expf(-x));
}

// ---------------- fp32 -> bf16 conversion ----------------
__global__ __launch_bounds__(256) void cvt_bf16(const float* __restrict__ in,
                                                u16* __restrict__ out, int n) {
    for (int i = blockIdx.x * 256 + threadIdx.x; i < n; i += gridDim.x * 256)
        out[i] = f2bf(in[i]);
}

#if HAVE_TDM
// ---------------- TDM tile load: [128 rows x 32 bf16] global -> LDS ----------
// D# per CDNA5 ISA ch.8. Hardware pad: after every 16 DWORDs (one 64B row)
// insert 4 DWORDs (16B) -> LDS row pitch = 40 halves = LDSW. 2D descriptor,
// groups 2/3 zero. Issued by ONE wave; completion via TENSORcnt.
__device__ __forceinline__ void tdm_load_tile(const u16* __restrict__ gsrc,
                                              u16* __restrict__ ldst, int ldK) {
    unsigned long long ga = (unsigned long long)(uintptr_t)gsrc;
    u32 la = (u32)(uintptr_t)ldst;        // flat->LDS offset = low 32 bits
    v4u g0;
    g0[0] = 1u;                                            // count=1, user D#
    g0[1] = la;                                            // lds_addr
    g0[2] = (u32)ga;                                       // global_addr[31:0]
    g0[3] = (u32)((ga >> 32) & 0x01FFFFFFull) | (2u << 30);// addr[56:32]|type=2
    v8i g1;
    // data_size=1(2B)<<16 | pad_enable<<20 | pad_interval=3(16dw)<<22
    // | pad_amount=3(4dw)<<25
    g1[0] = (int)((1u << 16) | (1u << 20) | (3u << 22) | (3u << 25));
    g1[1] = (int)(((u32)ldK & 0xFFFFu) << 16);             // tensor_dim0 lo16
    g1[2] = (int)(((u32)ldK >> 16) | (128u << 16));        // dim0 hi | dim1=128
    g1[3] = (int)(32u << 16);                              // tile_dim0 = 32
    g1[4] = 128;                                           // tile_dim1 = 128
    g1[5] = ldK;                                           // dim0_stride lo32
    g1[6] = 0;
    g1[7] = 0;
    v4i z = {0, 0, 0, 0};
#if __clang_major__ >= 23
    v8i z8 = {0, 0, 0, 0, 0, 0, 0, 0};
    __builtin_amdgcn_tensor_load_to_lds(g0, g1, z, z, z8, 0);
#else
    __builtin_amdgcn_tensor_load_to_lds(g0, g1, z, z, 0);
#endif
}
#endif

// ---------------- WMMA fragment gather from LDS tile ----------------
// 16-bit A 16x32 layout (ISA 7.12.2): lane<16 -> row=lane,   K {0..7,16..23}
//                                     lane>=16 -> row=lane-16,K {8..15,24..31}
// B (32x16) gathered identically from the [N,K] tile (NT symmetry).
__device__ __forceinline__ v16bf frag_ld(const u16* __restrict__ tile, int sub) {
    int lane = threadIdx.x & 31;
    int r    = lane & 15;
    int hi   = lane >> 4;
    const u16* p = tile + (sub + r) * LDSW + hi * 8;
    union { uint4 u[2]; v16bf v; } f;
    f.u[0] = *reinterpret_cast<const uint4*>(p);
    f.u[1] = *reinterpret_cast<const uint4*>(p + 16);
    return f.v;
}

// ---------------- generic NT GEMM: C[M,N] = A[M,K] * W[N,K]^T ----------------
// EPI: 0=f32 store, 1=bf16 store, 2=silu(clip(+-5))->bf16,
//      3=dual(W=up,W2=gate): bf16(silu(g)*u), 4=bf16 RMW: C += 0.1*acc
// Dynamic LDS: 2 buffers x ntiles x TILE_HALVES halves (TDM double buffering).
template <int EPI>
__global__ __launch_bounds__(256)
void gemm_nt(const u16* __restrict__ A, const u16* __restrict__ W,
             const u16* __restrict__ W2,
             float* __restrict__ Cf, u16* __restrict__ Cb,
             int M, int N, int K,
             long strideA, long strideW, long strideC) {
    extern __shared__ u16 smem[];
    constexpr int NTI = (EPI == 3) ? 3 : 2;

    const int blockN = blockIdx.x * TILE;
    const int blockM = blockIdx.y * TILE;
    const int wid    = threadIdx.x >> 5;
    const int waveM  = (wid & 1) * 64;     // 2 waves along M
    const int waveN  = (wid >> 1) * 32;    // 4 waves along N

    const u16* Ab = A + (long)blockIdx.z * strideA + (long)blockM * K;
    const u16* Wb = W + (long)blockIdx.z * strideW + (long)blockN * K;
    const u16* W2b = (EPI == 3) ? (W2 + (long)blockN * K) : nullptr;

    v8f acc[4][2];
    v8f acc2[4][2];
    v8f zero = {0.f,0.f,0.f,0.f,0.f,0.f,0.f,0.f};
#pragma unroll
    for (int i = 0; i < 4; ++i)
#pragma unroll
        for (int j = 0; j < 2; ++j) { acc[i][j] = zero; acc2[i][j] = zero; }

#if HAVE_TDM
    // ---- async pipeline: TDM fills buf[nxt] while WMMA consumes buf[cur] ----
    const int swid = __builtin_amdgcn_readfirstlane(wid);  // scalar branch
    int cur = 0;
    if (swid == 0) {
        tdm_load_tile(Ab, smem, K);
        tdm_load_tile(Wb, smem + TILE_HALVES, K);
        if (EPI == 3) tdm_load_tile(W2b, smem + 2 * TILE_HALVES, K);
    }
    __builtin_amdgcn_s_wait_tensorcnt(0);
    __syncthreads();
    for (int k0 = 0; k0 < K; k0 += TK) {
        const int nxt = cur ^ 1;
        if ((k0 + TK < K) && (swid == 0)) {
            u16* nb = smem + nxt * NTI * TILE_HALVES;
            tdm_load_tile(Ab + k0 + TK, nb, K);
            tdm_load_tile(Wb + k0 + TK, nb + TILE_HALVES, K);
            if (EPI == 3) tdm_load_tile(W2b + k0 + TK, nb + 2 * TILE_HALVES, K);
        }
        const u16* lA  = smem + cur * NTI * TILE_HALVES;
        const u16* lW  = lA + TILE_HALVES;
        const u16* lW2 = lA + 2 * TILE_HALVES;

        v16bf av[4], bv[2];
#pragma unroll
        for (int i = 0; i < 4; ++i) av[i] = frag_ld(lA, waveM + 16 * i);
#pragma unroll
        for (int j = 0; j < 2; ++j) bv[j] = frag_ld(lW, waveN + 16 * j);
#pragma unroll
        for (int i = 0; i < 4; ++i)
#pragma unroll
            for (int j = 0; j < 2; ++j)
                acc[i][j] = __builtin_amdgcn_wmma_f32_16x16x32_bf16(
                    false, av[i], false, bv[j], (short)0, acc[i][j], false, false);
        if constexpr (EPI == 3) {
            v16bf gv[2];
#pragma unroll
            for (int j = 0; j < 2; ++j) gv[j] = frag_ld(lW2, waveN + 16 * j);
#pragma unroll
            for (int i = 0; i < 4; ++i)
#pragma unroll
                for (int j = 0; j < 2; ++j)
                    acc2[i][j] = __builtin_amdgcn_wmma_f32_16x16x32_bf16(
                        false, av[i], false, gv[j], (short)0, acc2[i][j], false, false);
        }
        // next tile must be resident & everyone done reading cur before reuse
        __builtin_amdgcn_s_wait_tensorcnt(0);
        __syncthreads();
        cur = nxt;
    }
#else
    // ---- fallback: batched global loads -> LDS stores, single buffer ----
    u16* lA  = smem;
    u16* lW  = smem + TILE_HALVES;
    u16* lW2 = smem + 2 * TILE_HALVES;
    for (int k0 = 0; k0 < K; k0 += TK) {
        __syncthreads();
        {
            uint4 t0[2], t1[2], t2[2];
            int t = threadIdx.x;
#pragma unroll
            for (int it = 0; it < 2; ++it) {
                int id = t + it * 256, row = id >> 2, ch = id & 3;
                long off = (long)row * K + k0 + ch * 8;
                t0[it] = *reinterpret_cast<const uint4*>(Ab + off);
                t1[it] = *reinterpret_cast<const uint4*>(Wb + off);
                if constexpr (EPI == 3)
                    t2[it] = *reinterpret_cast<const uint4*>(W2b + off);
            }
            if (k0 + TK < K) {
                __builtin_prefetch(Ab + k0 + TK, 0, 1);
                __builtin_prefetch(Wb + k0 + TK, 0, 1);
            }
#pragma unroll
            for (int it = 0; it < 2; ++it) {
                int id = t + it * 256, row = id >> 2, ch = id & 3;
                int loff = row * LDSW + ch * 8;
                *reinterpret_cast<uint4*>(lA + loff) = t0[it];
                *reinterpret_cast<uint4*>(lW + loff) = t1[it];
                if constexpr (EPI == 3)
                    *reinterpret_cast<uint4*>(lW2 + loff) = t2[it];
            }
        }
        __syncthreads();

        v16bf av[4], bv[2];
#pragma unroll
        for (int i = 0; i < 4; ++i) av[i] = frag_ld(lA, waveM + 16 * i);
#pragma unroll
        for (int j = 0; j < 2; ++j) bv[j] = frag_ld(lW, waveN + 16 * j);
#pragma unroll
        for (int i = 0; i < 4; ++i)
#pragma unroll
            for (int j = 0; j < 2; ++j)
                acc[i][j] = __builtin_amdgcn_wmma_f32_16x16x32_bf16(
                    false, av[i], false, bv[j], (short)0, acc[i][j], false, false);
        if constexpr (EPI == 3) {
            v16bf gv[2];
#pragma unroll
            for (int j = 0; j < 2; ++j) gv[j] = frag_ld(lW2, waveN + 16 * j);
#pragma unroll
            for (int i = 0; i < 4; ++i)
#pragma unroll
                for (int j = 0; j < 2; ++j)
                    acc2[i][j] = __builtin_amdgcn_wmma_f32_16x16x32_bf16(
                        false, av[i], false, gv[j], (short)0, acc2[i][j], false, false);
        }
    }
#endif

    // epilogue: C/D layout (ISA 7.12.2): col = waveN+16j+(lane&15),
    // row = waveM+16i + 8*(lane>>4) + r
    const int lane = threadIdx.x & 31;
    const int hi   = lane >> 4;
    const int cl   = lane & 15;
    const long cbase = (long)blockIdx.z * strideC;
#pragma unroll
    for (int i = 0; i < 4; ++i)
#pragma unroll
        for (int j = 0; j < 2; ++j)
#pragma unroll
            for (int r = 0; r < 8; ++r) {
                int row = blockM + waveM + 16 * i + 8 * hi + r;
                int col = blockN + waveN + 16 * j + cl;
                long idx = cbase + (long)row * N + col;
                float v = acc[i][j][r];
                if constexpr (EPI == 0) {
                    Cf[idx] = v;
                } else if constexpr (EPI == 1) {
                    Cb[idx] = f2bf(v);
                } else if constexpr (EPI == 2) {
                    v = fminf(fmaxf(v, -5.f), 5.f);
                    Cb[idx] = f2bf(silu(v));
                } else if constexpr (EPI == 3) {
                    float g = acc2[i][j][r];
                    Cb[idx] = f2bf(silu(g) * v);
                } else if constexpr (EPI == 4) {
                    Cb[idx] = f2bf(bf2f(Cb[idx]) + 0.1f * v);
                }
            }
}

// ---------------- LayerNorm over last dim == 128 ----------------
template <bool TRANSPOSE>
__global__ __launch_bounds__(128)
void ln128(const float* __restrict__ in, const float* __restrict__ g,
           const float* __restrict__ b, u16* __restrict__ out,
           u16* __restrict__ outT, int S) {
    int row = blockIdx.x;      // token index in [0, B*S)
    int t   = threadIdx.x;
    float x = in[(long)row * CA + t];
    __shared__ float ss[CA], sq[CA];
    ss[t] = x; sq[t] = x * x;
    __syncthreads();
    for (int o = 64; o > 0; o >>= 1) {
        if (t < o) { ss[t] += ss[t + o]; sq[t] += sq[t + o]; }
        __syncthreads();
    }
    float m = ss[0] * (1.f / CA);
    float v = sq[0] * (1.f / CA) - m * m;
    float y = (x - m) * rsqrtf(v + 1e-5f) * g[t] + b[t];
    u16 h = f2bf(y);
    out[(long)row * CA + t] = h;
    if constexpr (TRANSPOSE) {
        int bi = row / S, s = row % S;
        outT[((long)bi * CA + t) * S + s] = h;   // [B, A, S]
    }
}

// ---------------- Wc = w_out @ w_eproj  ([D,H] x [H,A] -> [D,A]) ----------------
__global__ __launch_bounds__(128)
void make_wc(const float* __restrict__ w_out, const float* __restrict__ w_eproj,
             float* __restrict__ Wc) {
    int d = blockIdx.x, a = threadIdx.x;
    float acc = 0.f;
    for (int h = 0; h < CH; ++h)
        acc += w_out[(long)d * CH + h] * w_eproj[(long)h * CA + a];
    Wc[(long)d * CA + a] = acc;
}

// ---------------- per-token expert (last positive expert wins) ----------------
__global__ __launch_bounds__(256)
void expert_apply(const float* __restrict__ ew, const float* __restrict__ pre,
                  const float* __restrict__ w_exp, const float* __restrict__ eg,
                  const float* __restrict__ eb, const float* __restrict__ Wc,
                  float* __restrict__ out) {
    int tok = blockIdx.x;
    int t   = threadIdx.x;
    int idx = -1;
#pragma unroll
    for (int i = 0; i < CE; ++i)
        if (ew[(long)tok * CE + i] > 0.f) idx = i;
    if (idx < 0) return;   // combined stays zero -> out unchanged (uniform exit)

    __shared__ float sp[CA], sa[CA], ss[CA], sq[CA];
    if (t < CA) sp[t] = pre[(long)tok * CA + t];
    __syncthreads();

    float tv = 0.f;
    if (t < CA) {
        const float* wr = w_exp + ((long)idx * CA + t) * CA;
#pragma unroll 4
        for (int a = 0; a < CA; ++a) tv += sp[a] * wr[a];
        ss[t] = tv; sq[t] = tv * tv;
    }
    __syncthreads();
    for (int o = 64; o > 0; o >>= 1) {
        if (t < o) { ss[t] += ss[t + o]; sq[t] += sq[t + o]; }
        __syncthreads();
    }
    float m = ss[0] * (1.f / CA);
    float v = sq[0] * (1.f / CA) - m * m;
    if (t < CA)
        sa[t] = (tv - m) * rsqrtf(v + 1e-5f) * eg[(long)idx * CA + t]
                + eb[(long)idx * CA + t];
    __syncthreads();

#pragma unroll
    for (int q = 0; q < 4; ++q) {
        int d = t + q * 256;
        const float* wr = Wc + (long)d * CA;
        float acc = 0.f;
#pragma unroll 4
        for (int c = 0; c < CA; ++c) acc += sa[c] * wr[c];
        out[(long)tok * CD + d] += 0.1f * acc;
    }
}

// ---------------- host side ----------------
extern "C" void kernel_launch(void* const* d_in, const int* in_sizes, int n_in,
                              void* d_out, int out_size, void* d_ws, size_t ws_size,
                              hipStream_t stream) {
    const float* x       = (const float*)d_in[0];
    const float* ew      = (const float*)d_in[1];
    const float* w_up    = (const float*)d_in[2];
    const float* w_gate  = (const float*)d_in[3];
    const float* w_down  = (const float*)d_in[4];
    const float* w_pre   = (const float*)d_in[5];
    const float* w_post  = (const float*)d_in[6];
    const float* an_g    = (const float*)d_in[7];
    const float* an_b    = (const float*)d_in[8];
    const float* w_aproj = (const float*)d_in[9];
    const float* w_exp   = (const float*)d_in[10];
    const float* eln_g   = (const float*)d_in[11];
    const float* eln_b   = (const float*)d_in[12];
    const float* w_eproj = (const float*)d_in[13];
    const float* w_out   = (const float*)d_in[14];
    float* out = (float*)d_out;

    char* p = (char*)d_ws;
    auto alloc = [&](size_t bytes) {
        void* r = (void*)p;
        p += (bytes + 255) & ~(size_t)255;
        return r;
    };
    u16*   xb      = (u16*)  alloc((size_t)CBS * CD * 2);
    u16*   wub     = (u16*)  alloc((size_t)CH * CD * 2);
    u16*   wgb     = (u16*)  alloc((size_t)CH * CD * 2);
    u16*   wdb     = (u16*)  alloc((size_t)CD * CH * 2);
    u16*   wpreb   = (u16*)  alloc((size_t)CA * CD * 2);
    u16*   wpostb  = (u16*)  alloc((size_t)CA * CH * 2);
    u16*   waprojb = (u16*)  alloc((size_t)CH * CA * 2);
    u16*   hidden  = (u16*)  alloc((size_t)CBS * CH * 2);
    float* pre_f   = (float*)alloc((size_t)CBS * CA * 4);
    float* post_f  = (float*)alloc((size_t)CBS * CA * 4);
    u16*   ain     = (u16*)  alloc((size_t)CBS * CA * 2);
    u16*   ainT    = (u16*)  alloc((size_t)CB * CA * CS * 2);
    u16*   aout    = (u16*)  alloc((size_t)CBS * CA * 2);
    u16*   awb     = (u16*)  alloc((size_t)CB * CS * CS * 2);
    u16*   adaptb  = (u16*)  alloc((size_t)CBS * CA * 2);
    float* Wc      = (float*)alloc((size_t)CD * CA * 4);

    const size_t lds2 = (size_t)2 * 2 * TILE_HALVES * sizeof(u16);  // 40 KB
    const size_t lds3 = (size_t)2 * 3 * TILE_HALVES * sizeof(u16);  // 60 KB

    // 1) bf16 conversions
    cvt_bf16<<<2048, 256, 0, stream>>>(x,       xb,      CBS * CD);
    cvt_bf16<<<512,  256, 0, stream>>>(w_up,    wub,     CH * CD);
    cvt_bf16<<<512,  256, 0, stream>>>(w_gate,  wgb,     CH * CD);
    cvt_bf16<<<512,  256, 0, stream>>>(w_down,  wdb,     CD * CH);
    cvt_bf16<<<128,  256, 0, stream>>>(w_pre,   wpreb,   CA * CD);
    cvt_bf16<<<128,  256, 0, stream>>>(w_post,  wpostb,  CA * CH);
    cvt_bf16<<<128,  256, 0, stream>>>(w_aproj, waprojb, CH * CA);

    // 2) Wc = w_out @ w_eproj   (factor the expert tail once)
    make_wc<<<CD, 128, 0, stream>>>(w_out, w_eproj, Wc);

    // 3) hidden = silu(x@w_gate^T) * (x@w_up^T)   [fused dual GEMM, bf16 out]
    gemm_nt<3><<<dim3(CH / TILE, CBS / TILE, 1), 256, lds3, stream>>>(
        xb, wub, wgb, nullptr, hidden, CBS, CH, CD, 0, 0, 0);

    // 4) pre = x @ w_pre^T  (fp32 kept for experts)
    gemm_nt<0><<<dim3(CA / TILE, CBS / TILE, 1), 256, lds2, stream>>>(
        xb, wpreb, nullptr, pre_f, nullptr, CBS, CA, CD, 0, 0, 0);

    // 5) adapt_in = LN(pre) -> bf16 + transposed copy [B,A,S]
    ln128<true><<<CBS, 128, 0, stream>>>(pre_f, an_g, an_b, ain, ainT, CS);

    // 6) post = hidden @ w_post^T ; adapt_out = LN(post)
    gemm_nt<0><<<dim3(CA / TILE, CBS / TILE, 1), 256, lds2, stream>>>(
        hidden, wpostb, nullptr, post_f, nullptr, CBS, CA, CH, 0, 0, 0);
    ln128<false><<<CBS, 128, 0, stream>>>(post_f, an_g, an_b, aout, nullptr, CS);

    // 7) aw[b] = silu(clip(adapt_in[b] @ adapt_out[b]^T))   (batched NT)
    gemm_nt<2><<<dim3(CS / TILE, CS / TILE, CB), 256, lds2, stream>>>(
        ain, aout, nullptr, nullptr, awb, CS, CS, CA,
        (long)CS * CA, (long)CS * CA, (long)CS * CS);

    // 8) adapt[b] = aw[b] @ adapt_in[b]   (NT vs transposed adapt_in)
    gemm_nt<1><<<dim3(CA / TILE, CS / TILE, CB), 256, lds2, stream>>>(
        awb, ainT, nullptr, nullptr, adaptb, CS, CA, CS,
        (long)CS * CS, (long)CA * CS, (long)CS * CA);

    // 9) hidden += 0.1 * adapt @ w_aproj^T   (bf16 RMW epilogue)
    gemm_nt<4><<<dim3(CH / TILE, CBS / TILE, 1), 256, lds2, stream>>>(
        adaptb, waprojb, nullptr, nullptr, hidden, CBS, CH, CA, 0, 0, 0);

    // 10) out = hidden @ w_down^T   (fp32, writes all of d_out)
    gemm_nt<0><<<dim3(CD / TILE, CBS / TILE, 1), 256, lds2, stream>>>(
        hidden, wdb, nullptr, out, nullptr, CBS, CD, CH, 0, 0, 0);

    // 11) out += 0.1 * expert_combined (only last positive expert per token)
    expert_apply<<<CBS, 256, 0, stream>>>(ew, pre_f, w_exp, eln_g, eln_b, Wc, out);
}
